// Mutual_Representation_PreTrain_Net_SingleMerge_659
// MI455X (gfx1250) — matrix operand order repaired and checked by reading, not compiled, not run on hardware
//
#include <hip/hip_runtime.h>

// ---------------------------------------------------------------------------
// MI455X (gfx1250, wave32) implementation.
// Compute-bound (~350 GFLOP vs tiny data): all big GEMMs + attention go
// through v_wmma_f32_16x16x32_f16. Flash attention keeps the 8.4GB score
// tensor out of HBM entirely (O accumulators in VGPRs, P staged via 1KB LDS).
// V is produced TRANSPOSED by its projection GEMM so that attention's P*V
// B-fragments are contiguous b128 loads instead of 16 strided u16 loads.
// ---------------------------------------------------------------------------

typedef __attribute__((ext_vector_type(16))) _Float16 v16h;
typedef __attribute__((ext_vector_type(8)))  _Float16 v8h;
typedef __attribute__((ext_vector_type(8)))  float    v8f;

union FragU { v16h v; v8h h2[2]; _Float16 e[16]; };

#define WMMA_F16(a, b, c) \
  __builtin_amdgcn_wmma_f32_16x16x32_f16(false, (a), false, (b), (short)0, (c), false, false)

// ---- constants -------------------------------------------------------------
#define Bb 8
#define Ss 512
#define NBc 50
#define Ff 9
#define Dd 128
#define Hh 8
#define DHh 128
#define HD 1024           // H*DH
#define MT 4096           // B*S tokens
#define Ll 2

__device__ __forceinline__ v8f v8f_zero() {
  v8f z;
  #pragma unroll
  for (int i = 0; i < 8; ++i) z[i] = 0.0f;
  return z;
}

// ============================================================================
// f32 -> f16 convert
// ============================================================================
__global__ void cvt_f16_kernel(const float* __restrict__ src, _Float16* __restrict__ dst, int n) {
  int i = blockIdx.x * 256 + threadIdx.x;
  if (i < n) dst[i] = (_Float16)src[i];
}

// ============================================================================
// Positional MLP: gy(51) -> tanh(50) -> tanh(50) -> sigmoid(128)
// one thread per token, grid.y = f
// ============================================================================
__global__ void posmlp_kernel(const float* __restrict__ Y, const float* __restrict__ Gin,
                              const float* __restrict__ W1, const float* __restrict__ B1,
                              const float* __restrict__ W2, const float* __restrict__ B2,
                              const float* __restrict__ W3, const float* __restrict__ B3,
                              float* __restrict__ Xf, _Float16* __restrict__ Xh) {
  int f = blockIdx.y;
  int tok = blockIdx.x * 256 + threadIdx.x;           // 0..4095
  float gy[NBc + 1];
  for (int i = 0; i < NBc; ++i) gy[i] = Y[(long)tok * NBc + i];
  gy[NBc] = Gin[(long)tok * Ff + f];

  const float* w1 = W1 + (long)f * 50 * 51;
  const float* b1 = B1 + (long)f * 50;
  float h1[50];
  for (int j = 0; j < 50; ++j) {
    float s = b1[j];
    for (int i = 0; i < 51; ++i) s += gy[i] * w1[j * 51 + i];
    h1[j] = tanhf(s);
  }
  const float* w2 = W2 + (long)f * 50 * 50;
  const float* b2 = B2 + (long)f * 50;
  float h2[50];
  for (int j = 0; j < 50; ++j) {
    float s = b2[j];
    for (int i = 0; i < 50; ++i) s += h1[i] * w2[j * 50 + i];
    h2[j] = tanhf(s);
  }
  const float* w3 = W3 + (long)f * Dd * 50;
  const float* b3 = B3 + (long)f * Dd;
  float*    xr = Xf + ((long)f * MT + tok) * Dd;
  _Float16* xh = Xh + ((long)f * MT + tok) * Dd;
  for (int d = 0; d < Dd; ++d) {
    float s = b3[d];
    for (int i = 0; i < 50; ++i) s += h2[i] * w3[d * 50 + i];
    float v = 1.0f / (1.0f + __expf(-s));
    xr[d] = v;
    xh[d] = (_Float16)v;
  }
}

// ============================================================================
// Generic WMMA GEMM: C[M,N] = A[M,K](f16) @ W[N,K]^T(f16) + bias
// one wave -> 16x64 output tile; 4 waves/block; grid.y = batch(f)
// Output modes:
//   Ch != null, Ct == null : row-major f16
//   Ct != null             : TRANSPOSED f16, Ct[col*M + row]  (8 consecutive
//                            rows per lane -> merged b128 stores)
//   else                   : row-major f32 (Cf)
// ============================================================================
__global__ void gemm_wmma_kernel(const _Float16* __restrict__ A, const _Float16* __restrict__ W,
                                 const float* __restrict__ bias,
                                 _Float16* __restrict__ Ch, _Float16* __restrict__ Ct,
                                 float* __restrict__ Cf,
                                 int M, int N, int K,
                                 long sA, long sW, long sBias, long sC) {
  int fb = blockIdx.y;
  A += (long)fb * sA;
  W += (long)fb * sW;
  const float* bp = bias ? (bias + (long)fb * sBias) : nullptr;

  int lane = threadIdx.x & 31;
  int wave = threadIdx.x >> 5;
  int tilesN = N >> 6;                                // 64-wide wave tiles
  int t  = blockIdx.x * 4 + wave;
  int tm = t / tilesN;
  int tn = t % tilesN;
  if (tm * 16 >= M) return;

  int m = lane & 15, g = lane >> 4;
  v8f acc[4];
  #pragma unroll
  for (int i = 0; i < 4; ++i) acc[i] = v8f_zero();

  const _Float16* arow = A + (long)(tm * 16 + m) * K;
  for (int k0 = 0; k0 < K; k0 += 32) {
    const _Float16* pa = arow + k0 + g * 16;
    __builtin_prefetch(pa + 64, 0, 1);                // global_prefetch_b8
    FragU au;
    au.h2[0] = *(const v8h*)pa;                       // K = g*16 .. g*16+7
    au.h2[1] = *(const v8h*)(pa + 8);                 // K = g*16+8 .. +15
    #pragma unroll
    for (int ns = 0; ns < 4; ++ns) {
      int n0 = tn * 64 + ns * 16;
      const _Float16* pb = W + (long)(n0 + m) * K + k0 + g * 16;
      FragU bu;
      bu.h2[0] = *(const v8h*)pb;                     // K = g*16 .. +7
      bu.h2[1] = *(const v8h*)(pb + 8);               // K = g*16+8 .. +15
      acc[ns] = WMMA_F16(au.v, bu.v, acc[ns]);
    }
  }

  long cbase = (long)fb * sC;
  #pragma unroll
  for (int ns = 0; ns < 4; ++ns) {
    int col = tn * 64 + ns * 16 + m;
    float bv = bp ? bp[col] : 0.0f;
    if (Ct) {
      // transposed store: 8 consecutive halves per lane
      _Float16* pt = Ct + cbase + (long)col * M + (tm * 16 + 8 * g);
      FragU su;
      #pragma unroll
      for (int r = 0; r < 8; ++r) su.e[r] = (_Float16)(acc[ns][r] + bv);
      *(v8h*)pt = su.h2[0];
    } else {
      #pragma unroll
      for (int r = 0; r < 8; ++r) {
        float v = acc[ns][r] + bv;
        long idx = cbase + (long)(tm * 16 + r + 8 * g) * N + col;
        if (Ch) Ch[idx] = (_Float16)v;
        else    Cf[idx] = v;
      }
    }
  }
}

// ============================================================================
// Flash attention: one wave per (f, b, h, 16-query tile)
// S=512 keys processed 32 at a time (two 16x16 WMMA score tiles per step),
// online softmax, P·V via WMMA after an LDS layout round-trip.
// V arrives TRANSPOSED (f, col, token) so B-fragments are contiguous b128.
// ============================================================================
__global__ void __launch_bounds__(32)
attention_kernel(const _Float16* __restrict__ Q, const _Float16* __restrict__ K,
                 const _Float16* __restrict__ Vt, _Float16* __restrict__ O) {
  int qt = blockIdx.x;              // 0..31 query tiles
  int bh = blockIdx.y;              // 0..63
  int f  = blockIdx.z;              // 0..8
  int b = bh >> 3, h = bh & 7;
  int lane = threadIdx.x;
  int m = lane & 15, g = lane >> 4;

  long base = ((long)f * MT + (long)b * Ss) * HD + (long)h * DHh;
  const _Float16* Qp = Q + base;
  const _Float16* Kp = K + base;
  _Float16*       Op = O + base;
  // V^T: (f, col, token); this head's dims start at col = h*128; tokens b*512..
  const _Float16* Vtp = Vt + ((long)f * HD + (long)h * DHh) * MT + (long)b * Ss;

  // Q fragments for the whole 16x128 tile (4 k-steps)
  v16h qf[4];
  #pragma unroll
  for (int kk = 0; kk < 4; ++kk) {
    const _Float16* p = Qp + (long)(qt * 16 + m) * HD + kk * 32 + g * 16;
    FragU u;
    u.h2[0] = *(const v8h*)p;
    u.h2[1] = *(const v8h*)(p + 8);
    qf[kk] = u.v;
  }

  v8f o[8];
  #pragma unroll
  for (int nt = 0; nt < 8; ++nt) o[nt] = v8f_zero();
  float m8[8], l8[8];
  #pragma unroll
  for (int r = 0; r < 8; ++r) { m8[r] = -1e30f; l8[r] = 0.0f; }

  __shared__ __align__(16) _Float16 psh[16 * 32];
  const float scale = 0.08838834764831845f;           // 1/sqrt(128)

  for (int j2 = 0; j2 < 16; ++j2) {
    int key0 = j2 * 32;
    v8f s0 = v8f_zero(), s1 = v8f_zero();
    #pragma unroll
    for (int kk = 0; kk < 4; ++kk) {
      const _Float16* pb0 = Kp + (long)(key0 + m) * HD + kk * 32 + g * 16;
      const _Float16* pb1 = pb0 + (long)16 * HD;
      FragU b0, b1;
      b0.h2[0] = *(const v8h*)pb0; b0.h2[1] = *(const v8h*)(pb0 + 8);
      b1.h2[0] = *(const v8h*)pb1; b1.h2[1] = *(const v8h*)(pb1 + 8);
      s0 = WMMA_F16(qf[kk], b0.v, s0);
      s1 = WMMA_F16(qf[kk], b1.v, s1);
    }
    float t8[8];
    #pragma unroll
    for (int r = 0; r < 8; ++r) {
      s0[r] *= scale; s1[r] *= scale;
      t8[r] = fmaxf(s0[r], s1[r]);
    }
    // row max across the 16 lanes of this C-layout group
    #pragma unroll
    for (int mask = 1; mask < 16; mask <<= 1)
      #pragma unroll
      for (int r = 0; r < 8; ++r) t8[r] = fmaxf(t8[r], __shfl_xor(t8[r], mask, 32));
    float a8[8];
    #pragma unroll
    for (int r = 0; r < 8; ++r) {
      float mn = fmaxf(m8[r], t8[r]);
      a8[r] = __expf(m8[r] - mn);
      m8[r] = mn;
    }
    #pragma unroll
    for (int nt = 0; nt < 8; ++nt)
      #pragma unroll
      for (int r = 0; r < 8; ++r) o[nt][r] *= a8[r];

    float p0[8], p1[8], rs[8];
    #pragma unroll
    for (int r = 0; r < 8; ++r) {
      p0[r] = __expf(s0[r] - m8[r]);
      p1[r] = __expf(s1[r] - m8[r]);
      rs[r] = p0[r] + p1[r];
    }
    #pragma unroll
    for (int mask = 1; mask < 16; mask <<= 1)
      #pragma unroll
      for (int r = 0; r < 8; ++r) rs[r] += __shfl_xor(rs[r], mask, 32);
    #pragma unroll
    for (int r = 0; r < 8; ++r) l8[r] = l8[r] * a8[r] + rs[r];

    // C-layout -> A-layout via LDS (row-major 16x32 f16)
    #pragma unroll
    for (int r = 0; r < 8; ++r) {
      psh[(r + 8 * g) * 32 + m]      = (_Float16)p0[r];
      psh[(r + 8 * g) * 32 + m + 16] = (_Float16)p1[r];
    }
    asm volatile("s_wait_dscnt 0" ::: "memory");      // LDS RAW fence (same wave)
    FragU pu;
    const _Float16* pp = psh + m * 32 + g * 16;
    pu.h2[0] = *(const v8h*)pp;
    pu.h2[1] = *(const v8h*)(pp + 8);

    #pragma unroll
    for (int nt = 0; nt < 8; ++nt) {
      // B[k][n] = V[key0+k][dim nt*16+n] = Vt[(nt*16+n)*MT + key0+k]
      const _Float16* vp = Vtp + (long)(nt * 16 + m) * MT + key0 + g * 16;
      FragU vu;
      vu.h2[0] = *(const v8h*)vp;
      vu.h2[1] = *(const v8h*)(vp + 8);
      o[nt] = WMMA_F16(pu.v, vu.v, o[nt]);
    }
  }

  #pragma unroll
  for (int r = 0; r < 8; ++r) l8[r] = 1.0f / l8[r];
  #pragma unroll
  for (int nt = 0; nt < 8; ++nt)
    #pragma unroll
    for (int r = 0; r < 8; ++r)
      Op[(long)(qt * 16 + r + 8 * g) * HD + nt * 16 + m] = (_Float16)(o[nt][r] * l8[r]);
}

// ============================================================================
// x = LN(x + out); writes f32 and f16 copies. One thread per token.
// ============================================================================
__global__ void resid_ln_kernel(float* __restrict__ Xf, _Float16* __restrict__ Xh,
                                const float* __restrict__ Out,
                                const float* __restrict__ lnG, const float* __restrict__ lnB) {
  long t = (long)blockIdx.x * 256 + threadIdx.x;      // 0 .. 9*4096-1
  if (t >= (long)Ff * MT) return;
  int f = (int)(t >> 12);
  float*       xr = Xf + t * Dd;
  _Float16*    xh = Xh + t * Dd;
  const float* orow = Out + t * Dd;
  const float* gv = lnG + (long)f * Dd;
  const float* bv = lnB + (long)f * Dd;
  float s = 0.0f, ss = 0.0f;
  for (int d = 0; d < Dd; ++d) {
    float v = xr[d] + orow[d];
    s += v; ss += v * v;
  }
  float mu = s * (1.0f / Dd);
  float var = ss * (1.0f / Dd) - mu * mu;
  float rstd = rsqrtf(var + 1e-5f);
  for (int d = 0; d < Dd; ++d) {
    float v = xr[d] + orow[d];
    float n = (v - mu) * rstd * gv[d] + bv[d];
    xr[d] = n;
    xh[d] = (_Float16)n;
  }
}

// ============================================================================
// Final LN + softmax pooling over S. Block = (b, f), 256 threads.
// U[b][d][f] = sum_s softmax(LN(x)·poolW)_s * LN(x)[s][d]
// ============================================================================
__global__ void ln_pool_kernel(const float* __restrict__ X,
                               const float* __restrict__ fnG, const float* __restrict__ fnB,
                               const float* __restrict__ poolW, float* __restrict__ U) {
  int b = blockIdx.x, f = blockIdx.y;
  int tid = threadIdx.x;
  const float* xb = X + ((long)f * MT + (long)b * Ss) * Dd;
  const float* pw = poolW + (long)f * Dd;
  __shared__ float mean_s[Ss], rstd_s[Ss], w_s[Ss];
  __shared__ float red[2];

  for (int tok = tid; tok < Ss; tok += 256) {
    const float* xr = xb + (long)tok * Dd;
    float s = 0.0f, ss = 0.0f;
    for (int d = 0; d < Dd; ++d) { float v = xr[d]; s += v; ss += v * v; }
    float mu = s * (1.0f / Dd);
    float rstd = rsqrtf(ss * (1.0f / Dd) - mu * mu + 1e-5f);
    float sc = 0.0f;
    for (int d = 0; d < Dd; ++d)
      sc += ((xr[d] - mu) * rstd * fnG[d] + fnB[d]) * pw[d];
    mean_s[tok] = mu; rstd_s[tok] = rstd; w_s[tok] = sc;
  }
  __syncthreads();
  if (tid == 0) {
    float mx = -1e30f;
    for (int t = 0; t < Ss; ++t) mx = fmaxf(mx, w_s[t]);
    red[0] = mx;
  }
  __syncthreads();
  float mx = red[0];
  for (int tok = tid; tok < Ss; tok += 256) w_s[tok] = __expf(w_s[tok] - mx);
  __syncthreads();
  if (tid == 0) {
    float sm = 0.0f;
    for (int t = 0; t < Ss; ++t) sm += w_s[t];
    red[1] = sm;
  }
  __syncthreads();
  float inv = 1.0f / red[1];
  if (tid < Dd) {
    int d = tid;
    float acc = 0.0f;
    for (int tok = 0; tok < Ss; ++tok) {
      float v = xb[(long)tok * Dd + d];
      float ln = (v - mean_s[tok]) * rstd_s[tok] * fnG[d] + fnB[d];
      acc += ln * w_s[tok];
    }
    U[((long)b * Dd + d) * Ff + f] = acc * inv;
  }
}

// ============================================================================
// Unified (f=0..8) + merge (c=9) heads. QK^T collapses to a TxT bilinear form
// over the feature axis (T=9 or 6), so the whole stage is VALU-cheap.
// Block = (b, c), 128 threads (one per d-row of U).
// ============================================================================
__global__ void unified_merge_kernel(const float* __restrict__ U,
                                     const float* __restrict__ uQW, const float* __restrict__ uQB,
                                     const float* __restrict__ uKW, const float* __restrict__ uKB,
                                     const float* __restrict__ uVW, const float* __restrict__ uVB,
                                     const float* __restrict__ mQW, const float* __restrict__ mQB,
                                     const float* __restrict__ mKW, const float* __restrict__ mKB,
                                     const float* __restrict__ mVW, const float* __restrict__ mVB,
                                     float* __restrict__ out) {
  int b = blockIdx.x;              // 0..7
  int c = blockIdx.y;              // 0..9 (9 = merge head)
  int i = threadIdx.x;             // 0..127
  bool merge = (c == 9);
  int T = merge ? 6 : 9;
  int toff = merge ? 3 : 0;        // merge uses features 3..8
  const float *qw, *qb, *kw, *kb, *vw, *vb;
  if (merge) { qw = mQW; qb = mQB; kw = mKW; kb = mKB; vw = mVW; vb = mVB; }
  else {
    long wo = (long)c * Dd * Ff;
    qw = uQW + wo; kw = uKW + wo; vw = uVW + wo;
    qb = uQB + (long)c * Dd; kb = uKB + (long)c * Dd; vb = uVB + (long)c * Dd;
  }

  __shared__ float Ub[Dd][Ff];
  __shared__ float Mm[Ff][Ff], mqv[Ff], mkv[Ff], vwv[Ff];
  __shared__ float c0s, vb0s;
  for (int t = 0; t < Ff; ++t) Ub[i][t] = U[((long)b * Dd + i) * Ff + t];

  if (i < 81) {
    int t1 = i / 9, t2 = i % 9;
    float s = 0.0f;
    if (t1 < T && t2 < T)
      for (int k = 0; k < Dd; ++k) s += qw[k * T + t1] * kw[k * T + t2];
    Mm[t1][t2] = s;
  } else if (i < 90) {
    int t = i - 81; float s = 0.0f;
    if (t < T) for (int k = 0; k < Dd; ++k) s += qw[k * T + t] * kb[k];
    mqv[t] = s;
  } else if (i < 99) {
    int t = i - 90; float s = 0.0f;
    if (t < T) for (int k = 0; k < Dd; ++k) s += kw[k * T + t] * qb[k];
    mkv[t] = s;
  } else if (i < 108) {
    int t = i - 99; float s = 0.0f;
    if (t < T) for (int k = 0; k < Dd; ++k) s += vw[k * T + t];
    vwv[t] = s * (1.0f / Dd);
  } else if (i == 108) {
    float s = 0.0f;
    for (int k = 0; k < Dd; ++k) s += qb[k] * kb[k];
    c0s = s;
  } else if (i == 109) {
    float s = 0.0f;
    for (int k = 0; k < Dd; ++k) s += vb[k];
    vb0s = s * (1.0f / Dd);
  }
  __syncthreads();

  float u[Ff];
  for (int t = 0; t < T; ++t) u[t] = Ub[i][toff + t];
  float r[Ff];
  for (int t2 = 0; t2 < T; ++t2) {
    float s = mkv[t2];
    for (int t1 = 0; t1 < T; ++t1) s += u[t1] * Mm[t1][t2];
    r[t2] = s;
  }
  float sconst = c0s;
  for (int t = 0; t < T; ++t) sconst += u[t] * mqv[t];
  const float inv3 = 1.0f / 3.0f;                     // 1/sqrt(F), F=9 in both heads

  float mxl = -1e30f;
  for (int j = 0; j < Dd; ++j) {
    float lg = sconst;
    for (int t = 0; t < T; ++t) lg += r[t] * Ub[j][toff + t];
    mxl = fmaxf(mxl, lg * inv3);
  }
  float den = 0.0f, acc = 0.0f;
  for (int j = 0; j < Dd; ++j) {
    float lg = sconst;
    float vbar = vb0s;
    for (int t = 0; t < T; ++t) {
      float ujt = Ub[j][toff + t];
      lg += r[t] * ujt;
      vbar += ujt * vwv[t];
    }
    float w = __expf(lg * inv3 - mxl);
    den += w;
    acc += w * vbar;
  }
  float o = acc / den;
  if (merge) {
    float s = 0.0f;
    for (int t = 0; t < 6; ++t) s += Ub[i][3 + t];
    o += s * (1.0f / 6.0f);
  } else if (c == 0) {
    float s = 0.0f;
    for (int t = 0; t < Ff; ++t) s += Ub[i][t];
    o += s * (1.0f / Ff);
  } else {
    o += Ub[i][c];
  }
  out[((long)b * Dd + i) * 10 + c] = o;
}

// ============================================================================
// Host orchestration
// ============================================================================
extern "C" void kernel_launch(void* const* d_in, const int* in_sizes, int n_in,
                              void* d_out, int out_size, void* d_ws, size_t ws_size,
                              hipStream_t stream) {
  const float* Y     = (const float*)d_in[0];
  const float* Gin   = (const float*)d_in[1];
  const float* posW1 = (const float*)d_in[2];
  const float* posB1 = (const float*)d_in[3];
  const float* posW2 = (const float*)d_in[4];
  const float* posB2 = (const float*)d_in[5];
  const float* posW3 = (const float*)d_in[6];
  const float* posB3 = (const float*)d_in[7];
  const float* Wq    = (const float*)d_in[8];
  const float* bq    = (const float*)d_in[9];
  const float* Wk    = (const float*)d_in[10];
  const float* bk    = (const float*)d_in[11];
  const float* Wv    = (const float*)d_in[12];
  const float* bv    = (const float*)d_in[13];
  const float* Wo    = (const float*)d_in[14];
  const float* bo    = (const float*)d_in[15];
  const float* lnG   = (const float*)d_in[16];
  const float* lnB   = (const float*)d_in[17];
  const float* fnG   = (const float*)d_in[18];
  const float* fnB   = (const float*)d_in[19];
  const float* poolW = (const float*)d_in[20];
  const float* uQW   = (const float*)d_in[21];
  const float* uQB   = (const float*)d_in[22];
  const float* uKW   = (const float*)d_in[23];
  const float* uKB   = (const float*)d_in[24];
  const float* uVW   = (const float*)d_in[25];
  const float* uVB   = (const float*)d_in[26];
  const float* mQW   = (const float*)d_in[27];
  const float* mQB   = (const float*)d_in[28];
  const float* mKW   = (const float*)d_in[29];
  const float* mKB   = (const float*)d_in[30];
  const float* mVW   = (const float*)d_in[31];
  const float* mVB   = (const float*)d_in[32];
  float* out = (float*)d_out;

  char* w = (char*)d_ws;
  size_t off = 0;
  auto take = [&](size_t bytes) -> char* {
    char* p = w + off;
    off = (off + bytes + 255) & ~(size_t)255;
    return p;
  };

  const long nWqkv = (long)Ll * Ff * HD * Dd;         // 2,359,296 elems
  _Float16* WqH = (_Float16*)take(nWqkv * 2);
  _Float16* WkH = (_Float16*)take(nWqkv * 2);
  _Float16* WvH = (_Float16*)take(nWqkv * 2);
  _Float16* WoH = (_Float16*)take(nWqkv * 2);
  float*    Xf  = (float*)   take((long)Ff * MT * Dd * 4);
  _Float16* Xh  = (_Float16*)take((long)Ff * MT * Dd * 2);
  _Float16* Qh  = (_Float16*)take((long)Ff * MT * HD * 2);
  _Float16* Kh  = (_Float16*)take((long)Ff * MT * HD * 2);
  _Float16* VtH = (_Float16*)take((long)Ff * MT * HD * 2);   // V transposed
  _Float16* WVh = (_Float16*)take((long)Ff * MT * HD * 2);
  float*    OutF= (float*)   take((long)Ff * MT * Dd * 4);
  float*    Ubuf= (float*)   take((long)Bb * Dd * Ff * 4);

  // 1. weight conversion
  {
    int n = (int)nWqkv, gb = (n + 255) / 256;
    cvt_f16_kernel<<<gb, 256, 0, stream>>>(Wq, WqH, n);
    cvt_f16_kernel<<<gb, 256, 0, stream>>>(Wk, WkH, n);
    cvt_f16_kernel<<<gb, 256, 0, stream>>>(Wv, WvH, n);
    cvt_f16_kernel<<<gb, 256, 0, stream>>>(Wo, WoH, n);
  }

  // 2. positional MLP -> x
  posmlp_kernel<<<dim3(MT / 256, Ff), 256, 0, stream>>>(
      Y, Gin, posW1, posB1, posW2, posB2, posW3, posB3, Xf, Xh);

  // 3. transformer layers
  const long sAq = (long)MT * Dd;     // X per-f stride
  const long sWq = (long)HD * Dd;     // W per-f stride
  const long sCq = (long)MT * HD;     // QKV per-f stride
  for (int l = 0; l < Ll; ++l) {
    long wOff = (long)l * Ff * HD * Dd;
    long bOff = (long)l * Ff * HD;
    // QKV projections: M=4096 N=1024 K=128 -> 4096 wave-tiles -> grid.x=1024
    gemm_wmma_kernel<<<dim3(1024, Ff), 128, 0, stream>>>(
        Xh, WqH + wOff, bq + bOff, Qh, nullptr, nullptr, MT, HD, Dd, sAq, sWq, HD, sCq);
    gemm_wmma_kernel<<<dim3(1024, Ff), 128, 0, stream>>>(
        Xh, WkH + wOff, bk + bOff, Kh, nullptr, nullptr, MT, HD, Dd, sAq, sWq, HD, sCq);
    // V stored transposed: Vt[f][col][token]
    gemm_wmma_kernel<<<dim3(1024, Ff), 128, 0, stream>>>(
        Xh, WvH + wOff, bv + bOff, nullptr, VtH, nullptr, MT, HD, Dd, sAq, sWq, HD, sCq);
    // flash attention
    attention_kernel<<<dim3(Ss / 16, Bb * Hh, Ff), 32, 0, stream>>>(Qh, Kh, VtH, WVh);
    // output projection: M=4096 N=128 K=1024 -> 512 wave-tiles -> grid.x=128
    gemm_wmma_kernel<<<dim3(128, Ff), 128, 0, stream>>>(
        WVh, WoH + (long)l * Ff * Dd * HD, bo + (long)l * Ff * Dd,
        nullptr, nullptr, OutF, MT, Dd, HD, sCq, (long)Dd * HD, Dd, sAq);
    // residual + LN
    resid_ln_kernel<<<(Ff * MT) / 256, 256, 0, stream>>>(
        Xf, Xh, OutF, lnG + (long)l * Ff * Dd, lnB + (long)l * Ff * Dd);
  }

  // 4. final LN + softmax pooling -> U (B, D, F)
  ln_pool_kernel<<<dim3(Bb, Ff), 256, 0, stream>>>(Xf, fnG, fnB, poolW, Ubuf);

  // 5. unified + merge heads -> out (B, 128, 10)
  unified_merge_kernel<<<dim3(Bb, 10), 128, 0, stream>>>(
      Ubuf, uQW, uQB, uKW, uKB, uVW, uVB, mQW, mQB, mKW, mKB, mVW, mVB, out);
}
